// MultiHeadSelfAttention_90091234001321
// MI455X (gfx1250) — compile-verified, matrix operand order checked by reading, not run on hardware
//
#include <hip/hip_runtime.h>

typedef __attribute__((ext_vector_type(16))) __bf16 v16bf;
typedef __attribute__((ext_vector_type(8)))  __bf16 v8bf;
typedef __attribute__((ext_vector_type(8)))  float  v8f;

constexpr int Bc = 4, Sc = 2048, Dc = 1024, Hc = 16, DHc = 64;
constexpr int MT = Bc * Sc;            // 8192 total tokens
constexpr int LSTR = 72;               // LDS row stride in bf16 (16B-aligned, bank-skewed)

__device__ __forceinline__ v8f wmma_bf16(v16bf a, v16bf b, v8f c) {
  // D = A(16x32 bf16) x B(32x16 bf16) + C(16x16 f32)
  return __builtin_amdgcn_wmma_f32_16x16x32_bf16(false, a, false, b, (short)0, c, false, false);
}

// Load one 16x32 fragment line for this lane. 'line' must already include (lane&15).
// ISA A-layout: lanes 0-15 hold K{kb+0..7, kb+16..23}, lanes 16-31 K{kb+8..15, kb+24..31}.
__device__ __forceinline__ v16bf frag_ld(const __bf16* tile, int line, int kb, int lane) {
  const __bf16* p = tile + line * LSTR + kb + ((lane >> 4) << 3);
  v8bf lo = *(const v8bf*)(p);
  v8bf hi = *(const v8bf*)(p + 16);
  return __builtin_shufflevector(lo, hi, 0,1,2,3,4,5,6,7,8,9,10,11,12,13,14,15);
}

// ---------------- f32 -> bf16 conversion ----------------
__global__ __launch_bounds__(256) void cvt_f32_bf16(const float* __restrict__ in,
                                                    __bf16* __restrict__ out, int n) {
  int i = (blockIdx.x * 256 + threadIdx.x) * 8;
  if (i + 8 <= n) {
    float4 a = *(const float4*)(in + i);
    float4 b = *(const float4*)(in + i + 4);
    v8bf o;
    o[0] = (__bf16)a.x; o[1] = (__bf16)a.y; o[2] = (__bf16)a.z; o[3] = (__bf16)a.w;
    o[4] = (__bf16)b.x; o[5] = (__bf16)b.y; o[6] = (__bf16)b.z; o[7] = (__bf16)b.w;
    *(v8bf*)(out + i) = o;
  }
}

// ---------------- C[m,n] = sum_k A[m,k] * W[n,k]  (both K-contiguous) ----------------
// MODE 0: bf16 row-major [M,N];  MODE 1: f32 row-major;  MODE 2: bf16 V-transpose [B,H,DH,S]
template <int MODE>
__global__ __launch_bounds__(256) void gemm_nt_bf16(const __bf16* __restrict__ A,
                                                    const __bf16* __restrict__ W,
                                                    void* __restrict__ Cout,
                                                    int M, int N, int Kd) {
  __shared__ __bf16 sA[128 * LSTR];
  __shared__ __bf16 sB[128 * LSTR];
  const int tid = threadIdx.x, lane = tid & 31, wid = tid >> 5;
  const int wm = wid >> 2, wn = wid & 3;                 // 2x4 wave grid
  const int m0 = blockIdx.y * 128, n0 = blockIdx.x * 128;

  // staging geometry: thread handles 4 chunks (rows rbase + i*32, bytes kc*2)
  const int rbase = tid >> 3, kc = (tid & 7) << 3;
  const __bf16* pA = A + (size_t)(m0 + rbase) * Kd + kc;
  const __bf16* pW = W + (size_t)(n0 + rbase) * Kd + kc;
  __bf16* lA = sA + rbase * LSTR + kc;
  __bf16* lB = sB + rbase * LSTR + kc;

  v8f acc[4][2];
  for (int i = 0; i < 4; ++i) for (int j = 0; j < 2; ++j) acc[i][j] = (v8f)(0.f);

  // register double-buffer for global -> LDS
  v8bf gA[4], gB[4];
  for (int i = 0; i < 4; ++i) {
    gA[i] = *(const v8bf*)(pA + (size_t)i * 32 * Kd);
    gB[i] = *(const v8bf*)(pW + (size_t)i * 32 * Kd);
  }

  for (int k0 = 0; k0 < Kd; k0 += 64) {
    for (int i = 0; i < 4; ++i) {
      *(v8bf*)(lA + i * 32 * LSTR) = gA[i];
      *(v8bf*)(lB + i * 32 * LSTR) = gB[i];
    }
    __syncthreads();
    if (k0 + 64 < Kd) {      // prefetch next tile; latency hidden under the 16 WMMAs below
      for (int i = 0; i < 4; ++i) {
        gA[i] = *(const v8bf*)(pA + (size_t)i * 32 * Kd + k0 + 64);
        gB[i] = *(const v8bf*)(pW + (size_t)i * 32 * Kd + k0 + 64);
      }
    }
    for (int kk = 0; kk < 64; kk += 32) {
      v16bf af[4], bfr[2];
      for (int i = 0; i < 4; ++i) af[i]  = frag_ld(sA, wm * 64 + i * 16 + (lane & 15), kk, lane);
      for (int i = 0; i < 2; ++i) bfr[i] = frag_ld(sB, wn * 32 + i * 16 + (lane & 15), kk, lane);
      for (int mi = 0; mi < 4; ++mi)
        for (int ni = 0; ni < 2; ++ni)
          acc[mi][ni] = wmma_bf16(af[mi], bfr[ni], acc[mi][ni]);
    }
    __syncthreads();
  }

  // epilogue: C layout = VGPR j, lanes 0-15 -> M=j, lanes 16-31 -> M=j+8, N=lane&15
  const int mhalf = (lane >> 4) << 3;
  const int mb = m0 + wm * 64 + mhalf;
  const int nb = n0 + wn * 32 + (lane & 15);
  if (MODE == 0) {
    __bf16* base = (__bf16*)Cout + (size_t)mb * N + nb;
    for (int mi = 0; mi < 4; ++mi)
      for (int ni = 0; ni < 2; ++ni)
        for (int j = 0; j < 8; ++j)
          base[(size_t)(mi * 16 + j) * N + ni * 16] = (__bf16)acc[mi][ni][j];
  } else if (MODE == 1) {
    float* base = (float*)Cout + (size_t)mb * N + nb;
    for (int mi = 0; mi < 4; ++mi)
      for (int ni = 0; ni < 2; ++ni)
        for (int j = 0; j < 8; ++j)
          base[(size_t)(mi * 16 + j) * N + ni * 16] = acc[mi][ni][j];
  } else { // V transposed: [B,H,DH,S]
    for (int mi = 0; mi < 4; ++mi)
      for (int ni = 0; ni < 2; ++ni)
        for (int j = 0; j < 8; ++j) {
          int mr = mb + mi * 16 + j, nc = nb + ni * 16;
          int bb = mr >> 11, ss = mr & 2047;
          int hh = nc >> 6,  dh = nc & 63;
          ((__bf16*)Cout)[(((size_t)(bb * Hc + hh) * DHc + dh) << 11) + ss] = (__bf16)acc[mi][ni][j];
        }
  }
}

// ---------------- causal flash attention ----------------
// Q,K: bf16 [B,S,D] (head h at column h*64); Vt: bf16 [B,H,DH,S]; O: bf16 [B,S,D]
__global__ __launch_bounds__(256) void attn_fwd(const __bf16* __restrict__ Q,
                                                const __bf16* __restrict__ Kg,
                                                const __bf16* __restrict__ Vt,
                                                __bf16* __restrict__ O) {
  __shared__ __bf16 sK[64 * LSTR];           // 64 keys  x 64 dh
  __shared__ __bf16 sV[64 * LSTR];           // 64 dh    x 64 keys (from Vt)
  __shared__ __bf16 sP[8 * 16 * LSTR];       // per-wave P scratch (C-layout -> A-fragment)
  const int tid = threadIdx.x, lane = tid & 31, wid = tid >> 5;
  const int bh = blockIdx.y, b = bh >> 4, h = bh & 15;
  const int qt = blockIdx.x, q0 = qt * 128;
  const float scale2 = 0.125f * 1.44269504088896f;   // (1/sqrt(64)) * log2(e)
  const int mhalf = (lane >> 4) << 3;

  // staging geometry (2 chunks each for K and V tiles per thread)
  const int rbase = tid >> 3, kc = (tid & 7) << 3;
  const __bf16* pK = Kg + (size_t)(b * Sc + rbase) * Dc + h * DHc + kc;
  const __bf16* pV = Vt + (((size_t)(bh * DHc + rbase)) << 11) + kc;
  __bf16* lK = sK + rbase * LSTR + kc;
  __bf16* lV = sV + rbase * LSTR + kc;

  // Q fragments for this wave's 16 rows, kept in registers all kernel long
  const int qrow = q0 + wid * 16 + (lane & 15);
  const __bf16* qptr = Q + ((size_t)(b * Sc + qrow) * Dc + h * DHc);
  v16bf qA[2];
  for (int i = 0; i < 2; ++i) {
    const __bf16* p = qptr + i * 32 + mhalf;
    v8bf lo = *(const v8bf*)p, hi = *(const v8bf*)(p + 16);
    qA[i] = __builtin_shufflevector(lo, hi, 0,1,2,3,4,5,6,7,8,9,10,11,12,13,14,15);
  }

  v8f accO[4];
  for (int f = 0; f < 4; ++f) accO[f] = (v8f)(0.f);
  float rm[8], rl[8];
  for (int j = 0; j < 8; ++j) { rm[j] = -1e30f; rl[j] = 0.f; }

  __bf16* sPw = sP + wid * (16 * LSTR);

  const int nkv = 2 * qt + 2;

  // register double-buffer of the first KV tile
  v8bf gK[2], gV[2];
  for (int i = 0; i < 2; ++i) {
    gK[i] = *(const v8bf*)(pK + (size_t)(i * 32) * Dc);
    gV[i] = *(const v8bf*)(pV + ((size_t)(i * 32) << 11));
  }

  for (int t = 0; t < nkv; ++t) {
    const int kv0 = t * 64;
    for (int i = 0; i < 2; ++i) {
      *(v8bf*)(lK + i * 32 * LSTR) = gK[i];
      *(v8bf*)(lV + i * 32 * LSTR) = gV[i];
    }
    __syncthreads();
    if (t + 1 < nkv) {       // prefetch next KV tile under this tile's compute
      for (int i = 0; i < 2; ++i) {
        gK[i] = *(const v8bf*)(pK + (size_t)(kv0 + 64 + i * 32) * Dc);
        gV[i] = *(const v8bf*)(pV + ((size_t)(i * 32) << 11) + kv0 + 64);
      }
    }

    // S = Q * K^T   (A = Q rows, B columns = key rows, both dh-contiguous)
    v8f accS[4];
    for (int nf = 0; nf < 4; ++nf) accS[nf] = (v8f)(0.f);
    for (int kk = 0; kk < 64; kk += 32) {
      v16bf a = qA[kk >> 5];
      for (int nf = 0; nf < 4; ++nf) {
        v16bf bK = frag_ld(sK, nf * 16 + (lane & 15), kk, lane);
        accS[nf] = wmma_bf16(a, bK, accS[nf]);
      }
    }
    // scale (base-2 domain) + causal mask
    for (int nf = 0; nf < 4; ++nf) {
      int key = kv0 + nf * 16 + (lane & 15);
      for (int j = 0; j < 8; ++j) {
        int qm = q0 + wid * 16 + mhalf + j;
        float s = accS[nf][j] * scale2;
        accS[nf][j] = (key > qm) ? -1e30f : s;
      }
    }
    // online softmax: lanes 0-15 / 16-31 hold different rows -> reduce with masks 1,2,4,8 only
    for (int j = 0; j < 8; ++j) {
      float mx = fmaxf(fmaxf(accS[0][j], accS[1][j]), fmaxf(accS[2][j], accS[3][j]));
      for (int msk = 1; msk < 16; msk <<= 1) mx = fmaxf(mx, __shfl_xor(mx, msk, 32));
      float nm = fmaxf(rm[j], mx);
      float al = exp2f(rm[j] - nm);
      float rs = 0.f;
      for (int nf = 0; nf < 4; ++nf) {
        float p = exp2f(accS[nf][j] - nm);
        accS[nf][j] = p;
        rs += p;
      }
      for (int msk = 1; msk < 16; msk <<= 1) rs += __shfl_xor(rs, msk, 32);
      rl[j] = rl[j] * al + rs;
      rm[j] = nm;
      for (int f = 0; f < 4; ++f) accO[f][j] *= al;
    }
    // P: C-layout -> A-fragment layout via private LDS scratch (same-wave DS is in-order)
    for (int nf = 0; nf < 4; ++nf)
      for (int j = 0; j < 8; ++j)
        sPw[(mhalf + j) * LSTR + nf * 16 + (lane & 15)] = (__bf16)accS[nf][j];
    // O += P * V   (B columns = Vt rows, key-contiguous)
    for (int kk = 0; kk < 64; kk += 32) {
      v16bf aP = frag_ld(sPw, (lane & 15), kk, lane);
      for (int f = 0; f < 4; ++f) {
        v16bf bV = frag_ld(sV, f * 16 + (lane & 15), kk, lane);
        accO[f] = wmma_bf16(aP, bV, accO[f]);
      }
    }
    __syncthreads();
  }

  // epilogue: O / l, store bf16 into [B,S,D]
  for (int f = 0; f < 4; ++f) {
    int col = h * DHc + f * 16 + (lane & 15);
    for (int j = 0; j < 8; ++j) {
      int row = q0 + wid * 16 + mhalf + j;
      O[(size_t)(b * Sc + row) * Dc + col] = (__bf16)(accO[f][j] / rl[j]);
    }
  }
}

extern "C" void kernel_launch(void* const* d_in, const int* in_sizes, int n_in,
                              void* d_out, int out_size, void* d_ws, size_t ws_size,
                              hipStream_t stream) {
  (void)in_sizes; (void)n_in; (void)out_size; (void)ws_size;
  const float* x  = (const float*)d_in[0];
  const float* wq = (const float*)d_in[1];
  const float* wk = (const float*)d_in[2];
  const float* wv = (const float*)d_in[3];
  const float* wo = (const float*)d_in[4];
  float* out = (float*)d_out;

  // workspace layout (bf16 elements): ~88 MB total
  __bf16* ws  = (__bf16*)d_ws;
  size_t  off = 0;
  __bf16* xb  = ws + off; off += (size_t)MT * Dc;   // x bf16
  __bf16* wqb = ws + off; off += (size_t)Dc * Dc;
  __bf16* wkb = ws + off; off += (size_t)Dc * Dc;
  __bf16* wvb = ws + off; off += (size_t)Dc * Dc;
  __bf16* wob = ws + off; off += (size_t)Dc * Dc;
  __bf16* qb  = ws + off; off += (size_t)MT * Dc;   // Q [B,S,D]
  __bf16* kb  = ws + off; off += (size_t)MT * Dc;   // K [B,S,D]
  __bf16* vtb = ws + off; off += (size_t)MT * Dc;   // V^T [B,H,DH,S]
  __bf16* ab  = ws + off; off += (size_t)MT * Dc;   // attention output [B,S,D]

  const int nx = MT * Dc, nw = Dc * Dc;
  cvt_f32_bf16<<<nx / 2048, 256, 0, stream>>>(x,  xb,  nx);
  cvt_f32_bf16<<<nw / 2048, 256, 0, stream>>>(wq, wqb, nw);
  cvt_f32_bf16<<<nw / 2048, 256, 0, stream>>>(wk, wkb, nw);
  cvt_f32_bf16<<<nw / 2048, 256, 0, stream>>>(wv, wvb, nw);
  cvt_f32_bf16<<<nw / 2048, 256, 0, stream>>>(wo, wob, nw);

  dim3 gg(Dc / 128, MT / 128);
  gemm_nt_bf16<0><<<gg, 256, 0, stream>>>(xb, wqb, qb,  MT, Dc, Dc);
  gemm_nt_bf16<0><<<gg, 256, 0, stream>>>(xb, wkb, kb,  MT, Dc, Dc);
  gemm_nt_bf16<2><<<gg, 256, 0, stream>>>(xb, wvb, vtb, MT, Dc, Dc);

  attn_fwd<<<dim3(Sc / 128, Bc * Hc), 256, 0, stream>>>(qb, kb, vtb, ab);

  gemm_nt_bf16<1><<<gg, 256, 0, stream>>>(ab, wob, out, MT, Dc, Dc);
}
